// ResidualBlock_downsample_71193377898712
// MI455X (gfx1250) — matrix operand order repaired and checked by reading, not compiled, not run on hardware
//
#include <hip/hip_runtime.h>

// ---- problem constants (match reference) ----
#define N_IN    500000
#define N_OUT   125000
#define K_OFF   8
#define R_PAIRS 62500
#define CIN     64
#define COUT    128
#define BN_EPS  1e-4f
#define TILES   ((R_PAIRS + 15) / 16)       // 3907 row-tiles per offset (last tile padded)
#define TPB     16                          // row-tiles per block (weight reuse factor)
#define CHUNKS  ((TILES + TPB - 1) / TPB)   // 245 chunks per offset

// ---- workspace layout (bytes); total ~160.6 MB ----
#define OFF_X1    0ull            // 500000*64  bf16  = 64,000,000 B
#define OFF_Y1    64000000ull     // 125000*128 f32   = 64,000,000 B
#define OFF_Y1B   128000000ull    // 125000*128 bf16  = 32,000,000 B
#define OFF_W1B   160000000ull    // 8*64*128 bf16    = 131,072 B
#define OFF_W2B   160131072ull
#define OFF_WSB   160262144ull    // 8*128*128 bf16   = 262,144 B
#define OFF_STATS 160524288ull    // 16 KB stats

typedef __attribute__((ext_vector_type(16))) __bf16 v16bf;
typedef __attribute__((ext_vector_type(8)))  float  v8f;

__device__ __forceinline__ v8f wmma_bf16(v16bf a, v16bf b, v8f c) {
  // D = A(16x32 bf16) * B(32x16 bf16) + C(16x16 f32)
  return __builtin_amdgcn_wmma_f32_16x16x32_bf16(
      false, a, false, b, (short)0, c, false, false);
}

union FragBF { uint4 u[2]; v16bf v; };

// Async global->LDS 16B copy, GVS addressing (uniform SGPR base + 32-bit voffset),
// tracked by ASYNCcnt. CDNA5 path: no VGPR bounce, no per-lane 64-bit address math.
__device__ __forceinline__ void async_gather16(unsigned ldsByte, unsigned voff,
                                               const void* sbase) {
  asm volatile("global_load_async_to_lds_b128 %0, %1, %2"
               :: "v"(ldsByte), "v"(voff), "s"((unsigned long long)sbase) : "memory");
}
__device__ __forceinline__ void wait_async0() {
  asm volatile("s_wait_asynccnt 0x0" ::: "memory");
}

// ------------------------------------------------------------------
__global__ void zero_f32(float* __restrict__ p, int n) {
  int i = blockIdx.x * blockDim.x + threadIdx.x;
  int s = gridDim.x * blockDim.x;
  for (; i < n; i += s) p[i] = 0.0f;
}

// per-channel sum / sum-of-squares partials; float4 streaming (C divides 1024)
template <int C>
__global__ void bn_stats(const float* __restrict__ x, int nVec,
                         float* __restrict__ sums, float* __restrict__ sumsq) {
  __shared__ float sS[1024], sQ[1024];
  int t = threadIdx.x;
  float s0 = 0, s1 = 0, s2 = 0, s3 = 0, q0 = 0, q1 = 0, q2 = 0, q3 = 0;
  for (int i = blockIdx.x * 256 + t; i < nVec; i += gridDim.x * 256) {
    float4 v = ((const float4*)x)[i];
    s0 += v.x; q0 += v.x * v.x;
    s1 += v.y; q1 += v.y * v.y;
    s2 += v.z; q2 += v.z * v.z;
    s3 += v.w; q3 += v.w * v.w;
  }
  sS[4 * t + 0] = s0; sQ[4 * t + 0] = q0;
  sS[4 * t + 1] = s1; sQ[4 * t + 1] = q1;
  sS[4 * t + 2] = s2; sQ[4 * t + 2] = q2;
  sS[4 * t + 3] = s3; sQ[4 * t + 3] = q3;
  __syncthreads();
  if (t < C) {   // entries with index ≡ t (mod C) all belong to channel t
    float ts = 0.f, tq = 0.f;
    for (int g = t; g < 1024; g += C) { ts += sS[g]; tq += sQ[g]; }
    unsafeAtomicAdd(&sums[t], ts);
    unsafeAtomicAdd(&sumsq[t], tq);
  }
}

__global__ void bn_finalize(const float* __restrict__ sums, const float* __restrict__ sumsq,
                            const float* __restrict__ gamma, const float* __restrict__ beta,
                            float* __restrict__ scale, float* __restrict__ bias,
                            int C, float invN) {
  int c = threadIdx.x;
  if (c < C) {
    float m  = sums[c] * invN;
    float v  = sumsq[c] * invN - m * m;
    float sc = rsqrtf(v + BN_EPS) * gamma[c];
    scale[c] = sc;
    bias[c]  = beta[c] - m * sc;
  }
}

// y = relu(x*scale+bias) -> bf16, 8 elements/thread (2x b128 in, 1x b128 out)
template <int C>
__global__ void bn_apply(const float* __restrict__ x, const float* __restrict__ scale,
                         const float* __restrict__ bias, __bf16* __restrict__ out, int nVec) {
  int i = blockIdx.x * blockDim.x + threadIdx.x;
  int s = gridDim.x * blockDim.x;
  for (; i < nVec; i += s) {
    const float4* px = (const float4*)x + (size_t)i * 2;
    float4 lo = px[0], hi = px[1];
    int c0 = (i * 8) & (C - 1);          // C multiple of 8 -> 8 consecutive channels
    union { __bf16 h[8]; uint4 u; } o;
    float v;
    v = fmaf(lo.x, scale[c0 + 0], bias[c0 + 0]); o.h[0] = (__bf16)(v > 0.f ? v : 0.f);
    v = fmaf(lo.y, scale[c0 + 1], bias[c0 + 1]); o.h[1] = (__bf16)(v > 0.f ? v : 0.f);
    v = fmaf(lo.z, scale[c0 + 2], bias[c0 + 2]); o.h[2] = (__bf16)(v > 0.f ? v : 0.f);
    v = fmaf(lo.w, scale[c0 + 3], bias[c0 + 3]); o.h[3] = (__bf16)(v > 0.f ? v : 0.f);
    v = fmaf(hi.x, scale[c0 + 4], bias[c0 + 4]); o.h[4] = (__bf16)(v > 0.f ? v : 0.f);
    v = fmaf(hi.y, scale[c0 + 5], bias[c0 + 5]); o.h[5] = (__bf16)(v > 0.f ? v : 0.f);
    v = fmaf(hi.z, scale[c0 + 6], bias[c0 + 6]); o.h[6] = (__bf16)(v > 0.f ? v : 0.f);
    v = fmaf(hi.w, scale[c0 + 7], bias[c0 + 7]); o.h[7] = (__bf16)(v > 0.f ? v : 0.f);
    ((uint4*)out)[i] = o.u;
  }
}

// Prepack W[k][cin][cout] f32 -> per-lane WMMA B fragments (bf16), 32 B/lane contiguous.
__global__ void prepack_w(const float* __restrict__ W1, const float* __restrict__ W2,
                          const float* __restrict__ Ws,
                          __bf16* __restrict__ W1b, __bf16* __restrict__ W2b,
                          __bf16* __restrict__ Wsb) {
  int ti = blockIdx.x * blockDim.x + threadIdx.x;   // 512 frags * 32 lanes = 16384
  if (ti >= 16384) return;
  int f = ti >> 5, lane = ti & 31;
  const float* src; __bf16* dst; int cinTiles;
  if (f < 128)      { src = W1; dst = W1b; cinTiles = 2; }
  else if (f < 256) { f -= 128; src = W2; dst = W2b; cinTiles = 2; }
  else              { f -= 256; src = Ws; dst = Wsb; cinTiles = 4; }
  int perK = cinTiles * 8;
  int k = f / perK, rem = f % perK;
  int kt = rem >> 3, nt = rem & 7;
  int Cin = cinTiles * 32;
  int grp = lane >> 4, col = lane & 15;
  __bf16* d = dst + (size_t)f * 512 + (size_t)lane * 16;
  #pragma unroll
  for (int i = 0; i < 16; ++i) {
    int cin  = kt * 32 + grp * 16 + i;     // B layout: K = kt*32 + 16*(lane/16) + i
    int cout = nt * 16 + col;
    d[i] = (__bf16)src[((size_t)k * Cin + cin) * COUT + cout];
  }
}

// A-fragment from row-major bf16 LDS tile (contiguous K order, matches prepacked B):
// lane L: row=L%16, K = 16*(L/16) + i for i=0..15 (bytes {0..31}, +32B for lanes>=16).
__device__ __forceinline__ v16bf load_a(const __bf16* sA, int stride, int mrow, int grp, int kt) {
  FragBF A;
  const __bf16* p = sA + mrow * stride + kt * 32 + grp * 16;
  A.u[0] = *(const uint4*)(p);
  A.u[1] = *(const uint4*)(p + 8);
  return A.v;
}

__device__ __forceinline__ v16bf load_b(const __bf16* Wb, int fragIdx, int lane) {
  FragBF B;
  const uint4* p = (const uint4*)(Wb + (size_t)fragIdx * 512 + (size_t)lane * 16);
  B.u[0] = p[0];
  B.u[1] = p[1];
  return B.v;
}

// Fused conv1+conv2: block = (offset k, chunk of TPB row-tiles).
// B fragments live in VGPRs for the whole chunk; async double-buffered gather.
// Scatter uses uniform-base + 32-bit offset atomics; padded rows add 0.0f to row 0.
__global__ void __launch_bounds__(128) conv_dual(
    const __bf16* __restrict__ x1, const __bf16* __restrict__ W1b,
    const __bf16* __restrict__ W2b,
    const int* __restrict__ in_idx, const int* __restrict__ out_idx,
    float* __restrict__ y1, float* __restrict__ y2) {
  __shared__ __align__(16) __bf16 sbuf[2][16 * CIN];
  __shared__ int sOut[2][16];
  int k = blockIdx.x / CHUNKS;
  int t0 = (blockIdx.x % CHUNKS) * TPB;
  int ntiles = TILES - t0; if (ntiles > TPB) ntiles = TPB;
  int tid = threadIdx.x;
  int row = tid >> 3, seg = tid & 7;        // gather role: 16B per thread
  int wave = tid >> 5, lane = tid & 31;
  int grp = lane >> 4, mrow = lane & 15;
  int nt0 = wave * 2;
  unsigned coff = (unsigned)(nt0 * 16 + mrow);   // column of c?0; c?1 is +16

  // hoist all 8 B fragments (2 kt x 2 nt x {W1,W2}) into registers
  v16bf B10[2], B11[2], B20[2], B21[2];
  #pragma unroll
  for (int kt = 0; kt < 2; ++kt) {
    int fb = (k * 2 + kt) * 8;
    B10[kt] = load_b(W1b, fb + nt0, lane);
    B11[kt] = load_b(W1b, fb + nt0 + 1, lane);
    B20[kt] = load_b(W2b, fb + nt0, lane);
    B21[kt] = load_b(W2b, fb + nt0 + 1, lane);
  }

  auto issue = [&](int buf, int tile) {
    int r = tile * 16 + row; if (r >= R_PAIRS) r = R_PAIRS - 1;       // pad: clamp gather
    unsigned g = (unsigned)in_idx[k * R_PAIRS + r];
    unsigned lds = (unsigned)(unsigned long long)(const void*)&sbuf[buf][row * CIN]
                   + (unsigned)(seg * 16);
    async_gather16(lds, g * (CIN * 2) + (unsigned)(seg * 16), x1);
    if (tid < 16) {
      int rr = tile * 16 + tid;
      sOut[buf][tid] = (rr < R_PAIRS) ? out_idx[k * R_PAIRS + rr] * COUT : -1;
    }
  };

  issue(0, t0);                                    // prologue
  for (int i = 0; i < ntiles; ++i) {
    int cur = i & 1;
    wait_async0();                                 // own async stores to LDS done
    __syncthreads();                               // union over waves; prev reads done
    if (i + 1 < ntiles) issue(cur ^ 1, t0 + i + 1);

    v8f c10 = {}, c11 = {}, c20 = {}, c21 = {};
    #pragma unroll
    for (int kt = 0; kt < 2; ++kt) {
      v16bf a = load_a(sbuf[cur], CIN, mrow, grp, kt);
      c10 = wmma_bf16(a, B10[kt], c10);
      c11 = wmma_bf16(a, B11[kt], c11);
      c20 = wmma_bf16(a, B20[kt], c20);
      c21 = wmma_bf16(a, B21[kt], c21);
    }
    #pragma unroll
    for (int j = 0; j < 8; ++j) {                  // D[M][N]: M=j+8*grp, N=mrow
      int soff = sOut[cur][(grp << 3) + j];        // orow*COUT, or -1 for pad
      bool valid = soff >= 0;
      unsigned off = (valid ? (unsigned)soff : 0u) + coff;
      unsafeAtomicAdd(y1 + off,      valid ? c10[j] : 0.0f);
      unsafeAtomicAdd(y1 + off + 16, valid ? c11[j] : 0.0f);
      unsafeAtomicAdd(y2 + off,      valid ? c20[j] : 0.0f);
      unsafeAtomicAdd(y2 + off + 16, valid ? c21[j] : 0.0f);
    }
  }
}

// Submanifold conv 128->128: same scheme, 4 K-steps, 32B gather per thread.
__global__ void __launch_bounds__(128) conv_sub(
    const __bf16* __restrict__ y1b, const __bf16* __restrict__ Wsb,
    const int* __restrict__ in_idx, const int* __restrict__ out_idx,
    float* __restrict__ out) {
  __shared__ __align__(16) __bf16 sbuf[2][16 * COUT];
  __shared__ int sOut[2][16];
  int k = blockIdx.x / CHUNKS;
  int t0 = (blockIdx.x % CHUNKS) * TPB;
  int ntiles = TILES - t0; if (ntiles > TPB) ntiles = TPB;
  int tid = threadIdx.x;
  int row = tid >> 3, seg = tid & 7;        // 32B per thread
  int wave = tid >> 5, lane = tid & 31;
  int grp = lane >> 4, mrow = lane & 15;
  int nt0 = wave * 2;
  unsigned coff = (unsigned)(nt0 * 16 + mrow);

  v16bf B0[4], B1[4];
  #pragma unroll
  for (int kt = 0; kt < 4; ++kt) {
    int fb = (k * 4 + kt) * 8;
    B0[kt] = load_b(Wsb, fb + nt0, lane);
    B1[kt] = load_b(Wsb, fb + nt0 + 1, lane);
  }

  auto issue = [&](int buf, int tile) {
    int r = tile * 16 + row; if (r >= R_PAIRS) r = R_PAIRS - 1;
    unsigned g = (unsigned)in_idx[k * R_PAIRS + r];
    unsigned lds = (unsigned)(unsigned long long)(const void*)&sbuf[buf][row * COUT]
                   + (unsigned)(seg * 32);
    unsigned voff = g * (COUT * 2) + (unsigned)(seg * 32);
    async_gather16(lds,      voff,      y1b);
    async_gather16(lds + 16, voff + 16, y1b);
    if (tid < 16) {
      int rr = tile * 16 + tid;
      sOut[buf][tid] = (rr < R_PAIRS) ? out_idx[k * R_PAIRS + rr] * COUT : -1;
    }
  };

  issue(0, t0);
  for (int i = 0; i < ntiles; ++i) {
    int cur = i & 1;
    wait_async0();
    __syncthreads();
    if (i + 1 < ntiles) issue(cur ^ 1, t0 + i + 1);

    v8f c0 = {}, c1 = {};
    #pragma unroll
    for (int kt = 0; kt < 4; ++kt) {
      v16bf a = load_a(sbuf[cur], COUT, mrow, grp, kt);
      c0 = wmma_bf16(a, B0[kt], c0);
      c1 = wmma_bf16(a, B1[kt], c1);
    }
    #pragma unroll
    for (int j = 0; j < 8; ++j) {
      int soff = sOut[cur][(grp << 3) + j];
      bool valid = soff >= 0;
      unsigned off = (valid ? (unsigned)soff : 0u) + coff;
      unsafeAtomicAdd(out + off,      valid ? c0[j] : 0.0f);
      unsafeAtomicAdd(out + off + 16, valid ? c1[j] : 0.0f);
    }
  }
}

// ------------------------------------------------------------------
extern "C" void kernel_launch(void* const* d_in, const int* in_sizes, int n_in,
                              void* d_out, int out_size, void* d_ws, size_t ws_size,
                              hipStream_t stream) {
  const float* x      = (const float*)d_in[0];
  const float* gamma1 = (const float*)d_in[1];
  const float* beta1  = (const float*)d_in[2];
  const float* gamma2 = (const float*)d_in[3];
  const float* beta2  = (const float*)d_in[4];
  const float* W1     = (const float*)d_in[5];
  const float* Ws     = (const float*)d_in[6];
  const float* W2     = (const float*)d_in[7];
  const int* in_idx1  = (const int*)d_in[8];
  const int* out_idx1 = (const int*)d_in[9];
  const int* in_idx_s = (const int*)d_in[10];
  const int* out_idx_s= (const int*)d_in[11];
  float* out = (float*)d_out;

  char* ws = (char*)d_ws;
  __bf16* x1  = (__bf16*)(ws + OFF_X1);
  float*  y1  = (float*) (ws + OFF_Y1);
  __bf16* y1b = (__bf16*)(ws + OFF_Y1B);
  __bf16* W1b = (__bf16*)(ws + OFF_W1B);
  __bf16* W2b = (__bf16*)(ws + OFF_W2B);
  __bf16* Wsb = (__bf16*)(ws + OFF_WSB);
  float* st = (float*)(ws + OFF_STATS);
  float *sums1 = st,       *sumsq1 = st + 64,  *scale1 = st + 128, *bias1 = st + 192;
  float *sums2 = st + 256, *sumsq2 = st + 384, *scale2 = st + 512, *bias2 = st + 640;

  // zero accumulators / output / stats (harness poisons them)
  zero_f32<<<1024, 256, 0, stream>>>(out, N_OUT * COUT);
  zero_f32<<<1024, 256, 0, stream>>>(y1,  N_OUT * COUT);
  zero_f32<<<16,   256, 0, stream>>>(st,  4096);

  // BN1 + ReLU -> x1 (bf16)
  bn_stats<CIN><<<1024, 256, 0, stream>>>(x, (N_IN * CIN) / 4, sums1, sumsq1);
  bn_finalize<<<1, 64, 0, stream>>>(sums1, sumsq1, gamma1, beta1, scale1, bias1,
                                    CIN, 1.0f / (float)N_IN);
  bn_apply<CIN><<<2048, 256, 0, stream>>>(x, scale1, bias1, x1, (N_IN * CIN) / 8);

  // weights -> WMMA fragment layout (bf16)
  prepack_w<<<64, 256, 0, stream>>>(W1, W2, Ws, W1b, W2b, Wsb);

  // conv1 (-> y1) fused with conv2 (-> out), shared async gather
  conv_dual<<<K_OFF * CHUNKS, 128, 0, stream>>>(x1, W1b, W2b, in_idx1, out_idx1,
                                                y1, out);

  // BN2 + ReLU -> y1b (bf16)
  bn_stats<COUT><<<1024, 256, 0, stream>>>(y1, (N_OUT * COUT) / 4, sums2, sumsq2);
  bn_finalize<<<1, 128, 0, stream>>>(sums2, sumsq2, gamma2, beta2, scale2, bias2,
                                     COUT, 1.0f / (float)N_OUT);
  bn_apply<COUT><<<2048, 256, 0, stream>>>(y1, scale2, bias2, y1b, (N_OUT * COUT) / 8);

  // submanifold conv adds into out (residual AddTable: both convs += out)
  conv_sub<<<K_OFF * CHUNKS, 128, 0, stream>>>(y1b, Wsb, in_idx_s, out_idx_s,
                                               out);
}
